// InteractionNetwork_23313082483220
// MI455X (gfx1250) — matrix operand order repaired and checked by reading, not compiled, not run on hardware
//
#include <hip/hip_runtime.h>

typedef __attribute__((ext_vector_type(16))) _Float16 v16h;
typedef __attribute__((ext_vector_type(8)))  _Float16 v8h;
typedef __attribute__((ext_vector_type(8)))  float    v8f;

#define NOBJ 128   // objects
#define HR   150   // relational hidden width
#define HRP  160   // HR padded to multiple of 32 (K/N tile size)
#define KP   168   // LDS row stride in halves (bank-conflict pad, mult of 8)
#define DE   50    // effect dim
#define HO   100   // object hidden dim

// d_ws layout (f16 elements): W2h[160][168] | W3h[160][168] | W4h[64][168]
#define WS_W2_OFF 0
#define WS_W3_OFF (HRP * KP)
#define WS_W4_OFF (2 * HRP * KP)
#define WS_ELEMS  (2 * HRP * KP + 64 * KP)          // 64512 elems
#define WS_BYTES  (WS_ELEMS * 2)                    // 129024 bytes

struct SharedMem {
  _Float16 act[NOBJ][KP];     // activations (f16), wave-private rows
  _Float16 wbuf[HRP][KP];     // current layer weights [N][K] (f16, zero-padded)
  float Olds[5][NOBJ];        // O[b,:,:]
  float w1s[HR][5];           // Wr1 sender columns (cols 5..9)
  float c1[HRP];              // Wr1_rcv . O[:,o] + br1 (zero-padded)
  float effv[64];             // per-receiver effect accumulator
  float hobj[HO];             // object-MLP hidden
};

union FragU { v16h v; v8h h[2]; };

// ---- weight staging -------------------------------------------------------

// Fallback: fp32 global -> cvt -> LDS f16 (proven path).
template<int NT>
__device__ inline void stage_weights_f32(SharedMem& sm, const float* __restrict__ Wg,
                                         int Ncols, int tid)
{
  const int tot = NT * 16 * HRP;
  for (int e = tid; e < tot; e += 256) {
    int n = e / HRP, k = e - n * HRP;
    float w = (n < Ncols && k < HR) ? Wg[n * HR + k] : 0.0f;
    sm.wbuf[n][k] = (_Float16)w;
  }
}

// CDNA5 async path: weights pre-converted to the exact padded LDS image in
// d_ws; copy with GLOBAL_LOAD_ASYNC_TO_LDS_B128 (GVS mode: SGPR64 base +
// per-lane VGPR offset; VDST = LDS byte address). Tracked by ASYNCcnt.
template<int NT>
__device__ inline void stage_weights_async(SharedMem& sm, const _Float16* __restrict__ Wg16,
                                           int tid)
{
  // Low 32 bits of a generic LDS pointer are the LDS byte address (ISA 10.2).
  unsigned int lbase = (unsigned int)(unsigned long long)(&sm.wbuf[0][0]);
  unsigned long long gbase = (unsigned long long)Wg16;
  const int nbytes = NT * 16 * KP * 2;   // rows are fully padded in d_ws
  for (int c = tid * 16; c < nbytes; c += 256 * 16) {
    unsigned int laddr = lbase + (unsigned int)c;
    unsigned int goff  = (unsigned int)c;
    asm volatile("global_load_async_to_lds_b128 %0, %1, %2"
                 :: "v"(laddr), "v"(goff), "s"(gbase)
                 : "memory");
  }
  asm volatile("s_wait_asynccnt 0x0" ::: "memory");
}

// ---- one relational MLP layer --------------------------------------------
// act[128][0..160) @ W^T (+bias, ReLU). NT = number of 16-wide N tiles.
// LAST: reduce rows into effv instead of writing activations.
template<int NT, bool LAST, bool ASYNCW>
__device__ inline void relational_layer(SharedMem& sm,
                                        const float* __restrict__ Wg,
                                        const _Float16* __restrict__ Wg16,
                                        const float* __restrict__ bg,
                                        int Ncols, int tid)
{
  if (ASYNCW) stage_weights_async<NT>(sm, Wg16, tid);
  else        stage_weights_f32<NT>(sm, Wg, Ncols, tid);
  __syncthreads();  // weights staged; all waves done reading previous wbuf

  const int wv   = tid >> 5;    // wave id: owns rows [16*wv, 16*wv+16)
  const int lane = tid & 31;
  const int half = lane >> 4;   // lane 16..31 -> second K half
  const int l16  = lane & 15;
  const int r    = wv * 16 + l16;

  // Preload all A fragments (this wave's 16 rows, K = 0..160 in 5 steps of 32).
  // ISA 16-bit A 16x32 layout: lanes0-15 K {0..7,16..23}, lanes16-31 K {8..15,24..31}.
  FragU af[5];
#pragma unroll
  for (int kt = 0; kt < 5; ++kt) {
    int ka = kt * 32 + half * 8;
    af[kt].h[0] = *(const v8h*)&sm.act[r][ka];
    af[kt].h[1] = *(const v8h*)&sm.act[r][ka + 16];
  }

#pragma unroll
  for (int nt = 0; nt < NT; ++nt) {
    const int col = nt * 16 + l16;
    v8f acc = {};
#pragma unroll
    for (int kt = 0; kt < 5; ++kt) {
      // B 32x16 layout: lanes0-15 hold K {0..7,16..23}, lanes16-31 hold K {8..15,24..31} (matches A)
      FragU bfr;
      int kb = kt * 32 + half * 8;
      bfr.h[0] = *(const v8h*)&sm.wbuf[nt * 16 + l16][kb];
      bfr.h[1] = *(const v8h*)&sm.wbuf[nt * 16 + l16][kb + 16];
      acc = __builtin_amdgcn_wmma_f32_16x16x32_f16(false, af[kt].v, false, bfr.v,
                                                   (short)0, acc, false, false);
    }
    const float bias = (col < Ncols) ? bg[col] : 0.0f;
    if (!LAST) {
      // D layout: element v -> row 16*wv + 8*half + v, col = 16*nt + l16.
      // Safe to overwrite act: A-frags already register-resident, rows wave-private.
#pragma unroll
      for (int v = 0; v < 8; ++v) {
        int rowM = wv * 16 + half * 8 + v;
        sm.act[rowM][col] = (_Float16)fmaxf(acc[v] + bias, 0.0f);
      }
    } else {
      float part = 0.0f;
#pragma unroll
      for (int v = 0; v < 8; ++v) {
        int rowM = wv * 16 + half * 8 + v;
        float val = fmaxf(acc[v] + bias, 0.0f);
        part += (rowM == 127) ? 0.0f : val;   // mask the padding row
      }
      atomicAdd(&sm.effv[col], part);         // ds_add_f32
    }
  }
  __syncthreads();
}

// ---- phase 0: convert Wr2/Wr3/Wr4 to padded f16 LDS image in d_ws ---------
__global__ __launch_bounds__(256)
void prep_weights(const float* __restrict__ Wr2, const float* __restrict__ Wr3,
                  const float* __restrict__ Wr4, _Float16* __restrict__ ws)
{
  int idx = blockIdx.x * 256 + threadIdx.x;
  const int S2 = HRP * KP;
  if (idx < S2) {
    int n = idx / KP, k = idx - n * KP;
    ws[WS_W2_OFF + idx] = (_Float16)((n < HR && k < HR) ? Wr2[n * HR + k] : 0.0f);
  } else if (idx < 2 * S2) {
    int e = idx - S2; int n = e / KP, k = e - n * KP;
    ws[idx] = (_Float16)((n < HR && k < HR) ? Wr3[n * HR + k] : 0.0f);
  } else if (idx < WS_ELEMS) {
    int e = idx - 2 * S2; int n = e / KP, k = e - n * KP;
    ws[idx] = (_Float16)((n < DE && k < HR) ? Wr4[n * HR + k] : 0.0f);
  }
}

// ---- main fused Interaction Network kernel --------------------------------
template<bool ASYNCW>
__global__ __launch_bounds__(256)
void in_wmma_kernel(const float* __restrict__ O,
                    const float* __restrict__ Wr1, const float* __restrict__ br1,
                    const float* __restrict__ Wr2, const float* __restrict__ br2,
                    const float* __restrict__ Wr3, const float* __restrict__ br3,
                    const float* __restrict__ Wr4, const float* __restrict__ br4,
                    const float* __restrict__ Wo1, const float* __restrict__ bo1,
                    const float* __restrict__ Wo2, const float* __restrict__ bo2,
                    const _Float16* __restrict__ ws16,
                    float* __restrict__ out)
{
  extern __shared__ char smraw[];
  SharedMem& sm = *reinterpret_cast<SharedMem*>(smraw);
  const int tid = threadIdx.x;
  const int b = blockIdx.x >> 7;    // batch
  const int o = blockIdx.x & 127;   // receiver object

  // Stage O[b,:,:] and Wr1 sender columns; zero effect accumulator.
  for (int e = tid; e < 5 * NOBJ; e += 256)
    sm.Olds[e >> 7][e & 127] = O[(size_t)b * 5 * NOBJ + e];
  for (int e = tid; e < HR * 5; e += 256) {
    int h = e / 5, d = e - h * 5;
    sm.w1s[h][d] = Wr1[h * 11 + 5 + d];
  }
  if (tid < 64) sm.effv[tid] = 0.0f;
  __syncthreads();

  // c1[h] = br1[h] + sum_d Wr1[h,d] * O[b,d,o]  (receiver part, fp32 exact)
  for (int h = tid; h < HRP; h += 256) {
    float v = 0.0f;
    if (h < HR) {
      v = br1[h];
#pragma unroll
      for (int d = 0; d < 5; ++d) v += Wr1[h * 11 + d] * sm.Olds[d][o];
    }
    sm.c1[h] = v;
  }
  __syncthreads();

  // Layer 1 fused: h1[i][h] = relu(c1[h] + sum_d Wr1[h,5+d]*O[b,d,snd_i]).
  // Row i -> sender s = i + (i >= o); row 127 is zero padding.
  for (int e = tid; e < NOBJ * HRP; e += 256) {
    int i = e / HRP, h = e - i * HRP;
    float v = 0.0f;
    if (i != 127 && h < HR) {
      int s = i + (i >= o);
      float a = sm.c1[h];
#pragma unroll
      for (int d = 0; d < 5; ++d) a += sm.w1s[h][d] * sm.Olds[d][s];
      v = fmaxf(a, 0.0f);
    }
    sm.act[i][h] = (_Float16)v;
  }
  __syncthreads();

  // Layers 2..4 on WMMA (f16 in, fp32 accumulate).
  relational_layer<10, false, ASYNCW>(sm, Wr2, ws16 + WS_W2_OFF, br2, HR, tid);
  relational_layer<10, false, ASYNCW>(sm, Wr3, ws16 + WS_W3_OFF, br3, HR, tid);
  relational_layer<4,  true,  ASYNCW>(sm, Wr4, ws16 + WS_W4_OFF, br4, DE, tid);

  // Object MLP: C = [O[b,:,o] (5), eff (50)] -> 100 -> 2
  if (tid < HO) {
    float a = bo1[tid];
#pragma unroll
    for (int k = 0; k < 5; ++k) a += Wo1[tid * 55 + k] * sm.Olds[k][o];
    for (int k = 0; k < DE; ++k) a += Wo1[tid * 55 + 5 + k] * sm.effv[k];
    sm.hobj[tid] = fmaxf(a, 0.0f);
  }
  __syncthreads();
  if (tid < 2) {
    float a = bo2[tid];
    for (int j = 0; j < HO; ++j) a += Wo2[tid * HO + j] * sm.hobj[j];
    out[((size_t)b * 2 + tid) * NOBJ + o] = a;   // (B, D_p, N_O)
  }
}

extern "C" void kernel_launch(void* const* d_in, const int* in_sizes, int n_in,
                              void* d_out, int out_size, void* d_ws, size_t ws_size,
                              hipStream_t stream) {
  (void)in_sizes; (void)n_in; (void)out_size;
  const float* O   = (const float*)d_in[0];
  const float* Wr1 = (const float*)d_in[1];
  const float* br1 = (const float*)d_in[2];
  const float* Wr2 = (const float*)d_in[3];
  const float* br2 = (const float*)d_in[4];
  const float* Wr3 = (const float*)d_in[5];
  const float* br3 = (const float*)d_in[6];
  const float* Wr4 = (const float*)d_in[7];
  const float* br4 = (const float*)d_in[8];
  const float* Wo1 = (const float*)d_in[9];
  const float* bo1 = (const float*)d_in[10];
  const float* Wo2 = (const float*)d_in[11];
  const float* bo2 = (const float*)d_in[12];
  float* out = (float*)d_out;

  dim3 grid(64 * 128);   // one WG per (batch, receiver)
  dim3 block(256);       // 8 waves of 32

  if (ws_size >= (size_t)WS_BYTES) {
    _Float16* ws16 = (_Float16*)d_ws;
    prep_weights<<<(WS_ELEMS + 255) / 256, 256, 0, stream>>>(Wr2, Wr3, Wr4, ws16);
    in_wmma_kernel<true><<<grid, block, sizeof(SharedMem), stream>>>(
        O, Wr1, br1, Wr2, br2, Wr3, br3, Wr4, br4,
        Wo1, bo1, Wo2, bo2, (const _Float16*)ws16, out);
  } else {
    in_wmma_kernel<false><<<grid, block, sizeof(SharedMem), stream>>>(
        O, Wr1, br1, Wr2, br2, Wr3, br3, Wr4, br4,
        Wo1, bo1, Wo2, bo2, (const _Float16*)nullptr, out);
  }
}
